// DeformableConv_13640816132411
// MI455X (gfx1250) — matrix operand order repaired and checked by reading, not compiled, not run on hardware
//
#include <hip/hip_runtime.h>
#include <hip/hip_bf16.h>

typedef __attribute__((ext_vector_type(16))) __bf16       v16bf;
typedef __attribute__((ext_vector_type(8)))  float        v8f;
typedef __attribute__((ext_vector_type(8)))  unsigned int v8u;

#define Bn 4
#define Cc 256
#define Hh 64
#define Ww 64
#define Oo 256
#define HW 4096          // 64*64
#define KQ 2304          // C*9
#define KSTEPS 72        // KQ/32
#define NT 32            // spatial positions per tile
#define OFFS_BYTES 1179648u  // 4*18*4096*4

// Pack two fp32 -> two bf16 (RNE) in one dword with the CDNA5 packed convert.
// The clang builtin is not declared for gfx1250, so emit the instruction
// directly.
__device__ __forceinline__ unsigned int pack2bf(float lo, float hi) {
  unsigned int r;
  asm("v_cvt_pk_bf16_f32 %0, %1, %2" : "=v"(r) : "v"(lo), "v"(hi));
  return r;
}

// ---------------- Kernel 1: offset conv (3x3, C=256 -> 18ch) ----------------
__global__ void offs_conv(const float* __restrict__ x,
                          const float* __restrict__ w_off,
                          float* __restrict__ offs) {
  int idx = blockIdx.x * blockDim.x + threadIdx.x;     // 4*18*64*64 = 294912
  int w  = idx & 63;
  int h  = (idx >> 6) & 63;
  int t  = idx >> 12;         // b*18 + oc
  int oc = t % 18;
  int b  = t / 18;
  const float* xb = x + (size_t)b * Cc * HW;
  const float* wb = w_off + (size_t)oc * Cc * 9;
  float acc = 0.f;
  for (int c = 0; c < Cc; ++c) {
    const float* xp = xb + c * HW;
    const float* wp = wb + c * 9;
#pragma unroll
    for (int ky = 0; ky < 3; ++ky) {
      int y = h - 1 + ky;
      if ((unsigned)y >= 64u) continue;
#pragma unroll
      for (int kx = 0; kx < 3; ++kx) {
        int xc = w - 1 + kx;
        if ((unsigned)xc >= 64u) continue;
        acc = fmaf(xp[y * 64 + xc], wp[ky * 3 + kx], acc);
      }
    }
  }
  offs[idx] = acc;
}

// ------- Kernel 2: w_def (O,C,3,3) fp32 -> bf16 WMMA A-fragment order -------
// Layout: [mgroup 0..15][kstep 0..71][lane 0..31][8 x u32 (16 bf16)]
// 16-bit A layout: lane = {M = lane&15, khalf = lane>>4};
//   VGPR j: k = (j>=4?16:0) + khalf*8 + 2*(j&3) + {0,1}
// Global K index q = kstep*32 + k, with q = kk*256 + c (tap-major reorder).
__global__ void wfrag_build(const float* __restrict__ w_def,
                            unsigned int* __restrict__ wfrag) {
  int flat = blockIdx.x * blockDim.x + threadIdx.x;    // 16*72*32*8 = 294912
  int j    = flat & 7;
  int lane = (flat >> 3) & 31;
  int t2   = flat >> 8;
  int ks   = t2 % KSTEPS;
  int mg   = t2 / KSTEPS;
  int m     = mg * 16 + (lane & 15);
  int khalf = lane >> 4;
  int klo   = ((j >= 4) ? 16 : 0) + khalf * 8 + ((j & 3) << 1);
  float lo, hi;
  {
    int q = ks * 32 + klo;
    int kk = q >> 8, c = q & 255;
    lo = w_def[(size_t)m * KQ + c * 9 + kk];
  }
  {
    int q = ks * 32 + klo + 1;
    int kk = q >> 8, c = q & 255;
    hi = w_def[(size_t)m * KQ + c * 9 + kk];
  }
  wfrag[flat] = pack2bf(lo, hi);
}

// --------- Kernel 3: fused bilinear-gather + implicit GEMM via WMMA ---------
__global__ __launch_bounds__(512) void deform_wmma(
    const float* __restrict__ x,
    const float* __restrict__ offs,
    const unsigned int* __restrict__ wfrag,
    float* __restrict__ out) {
  // Per-(tap, position) bilinear params, fully boundary-resolved:
  //   p_idx: 4 pre-clamped flat spatial indices, packed as 4 x u16
  //   p_wgt: 4 bilinear weights with validity folded in (0 if OOB)
  __shared__ uint2  p_idx[9][NT];
  __shared__ float4 p_wgt[9][NT];
  // Ping-pong B-fragment staging: [buf][nhalf][(n&15)*32 + kp] bf16, kp = k with bits 3,4 swapped
  __shared__ __align__(32) unsigned short Vfrag[2][2][32 * 16];

  const int tid   = threadIdx.x;
  const int nbase = blockIdx.x * NT;          // 32 consecutive (b,h,w) positions
  const int b  = nbase >> 12;
  const int h  = (nbase >> 6) & 63;
  const int w0 = nbase & 63;

  // ---- precompute branch-free bilinear params ----
  if (tid < 9 * NT) {
    int kk = tid >> 5;
    int nl = tid & 31;
    int ky = kk / 3, kx = kk % 3;
    float dy = offs[(((size_t)b * 18 + kk * 2 + 0) * 64 + h) * 64 + (w0 + nl)];
    float dx = offs[(((size_t)b * 18 + kk * 2 + 1) * 64 + h) * 64 + (w0 + nl)];
    float py = (float)(h - 1 + ky) + dy;
    float px = (float)(w0 + nl - 1 + kx) + dx;
    float fy = floorf(py), fx = floorf(px);
    int y0 = (int)fy, x0 = (int)fx;
    int y1 = y0 + 1,  x1 = x0 + 1;
    float wy1 = py - fy, wx1 = px - fx;
    float wy0 = 1.f - wy1, wx0 = 1.f - wx1;
    bool yv0 = (unsigned)y0 < 64u, yv1 = (unsigned)y1 < 64u;
    bool xv0 = (unsigned)x0 < 64u, xv1 = (unsigned)x1 < 64u;
    int y0c = min(max(y0, 0), 63), y1c = min(max(y1, 0), 63);
    int x0c = min(max(x0, 0), 63), x1c = min(max(x1, 0), 63);
    unsigned i00 = (unsigned)(y0c * 64 + x0c), i01 = (unsigned)(y0c * 64 + x1c);
    unsigned i10 = (unsigned)(y1c * 64 + x0c), i11 = (unsigned)(y1c * 64 + x1c);
    uint2 pi;
    pi.x = i00 | (i01 << 16);
    pi.y = i10 | (i11 << 16);
    float4 pw;
    pw.x = (yv0 && xv0) ? wy0 * wx0 : 0.f;
    pw.y = (yv0 && xv1) ? wy0 * wx1 : 0.f;
    pw.z = (yv1 && xv0) ? wy1 * wx0 : 0.f;
    pw.w = (yv1 && xv1) ? wy1 * wx1 : 0.f;
    p_idx[kk][nl] = pi;
    p_wgt[kk][nl] = pw;
  }
  __syncthreads();

  const int lane    = tid & 31;
  const int wave    = tid >> 5;      // 0..15 -> M group (16 rows each)
  const int c_local = tid >> 4;      // 0..31 : channel within 32-wide K chunk
  const int n2      = (tid & 15) * 2;
  const int nhalf   = n2 >> 4;
  const float* xb   = x + (size_t)b * Cc * HW;

  v8f acc0 = {};
  v8f acc1 = {};

  for (int ks = 0; ks < KSTEPS; ++ks) {
    const int buf = ks & 1;
    // A-fragment load issued first so it overlaps the whole gather phase
    const v8u araw =
        *(const v8u*)(wfrag + (((size_t)wave * KSTEPS + ks) * 32 + lane) * 8);

    const int kk = ks >> 3;
    const int c  = ((ks & 7) << 5) + c_local;
    const float* xp = xb + (size_t)c * HW;

    // branchless bilinear gather for 2 positions
    const uint2  pi0 = p_idx[kk][n2];
    const float4 pw0 = p_wgt[kk][n2];
    const uint2  pi1 = p_idx[kk][n2 + 1];
    const float4 pw1 = p_wgt[kk][n2 + 1];
    float v0 = pw0.x * xp[pi0.x & 0xFFFFu] + pw0.y * xp[pi0.x >> 16]
             + pw0.z * xp[pi0.y & 0xFFFFu] + pw0.w * xp[pi0.y >> 16];
    float v1 = pw1.x * xp[pi1.x & 0xFFFFu] + pw1.y * xp[pi1.x >> 16]
             + pw1.z * xp[pi1.y & 0xFFFFu] + pw1.w * xp[pi1.y >> 16];
    unsigned int pack = pack2bf(v0, v1);

    const int kp = (c_local & 7) | ((c_local & 8) << 1) | ((c_local & 16) >> 1);
    Vfrag[buf][nhalf][(n2 & 15) * 32 + kp] = (unsigned short)(pack & 0xFFFFu);
    Vfrag[buf][nhalf][((n2 & 15) + 1) * 32 + kp] = (unsigned short)(pack >> 16);
    __syncthreads();

    v16bf a  = __builtin_bit_cast(v16bf, araw);
    v16bf b0 = __builtin_bit_cast(v16bf, *(const v8u*)&Vfrag[buf][0][(lane & 15) * 32 + (lane >> 4) * 16]);
    v16bf b1 = __builtin_bit_cast(v16bf, *(const v8u*)&Vfrag[buf][1][(lane & 15) * 32 + (lane >> 4) * 16]);
    acc0 = __builtin_amdgcn_wmma_f32_16x16x32_bf16(false, a, false, b0,
                                                   (short)0, acc0, false, false);
    acc1 = __builtin_amdgcn_wmma_f32_16x16x32_bf16(false, a, false, b1,
                                                   (short)0, acc1, false, false);
    // no trailing barrier: ping-pong buffer + the dscnt wait before each WMMA
    // make the next iteration's store into the other buffer race-free
  }

  // ---- epilogue: C/D layout (VGPR r, lanes 0-15: M=r, lanes 16-31: M=r+8) ----
  const int Mbase = wave * 16;
  const int nlo   = lane & 15;
  const int mofs  = (lane >> 4) * 8;
#pragma unroll
  for (int r = 0; r < 8; ++r) {
    int o = Mbase + mofs + r;
    size_t base = (((size_t)b * 256 + o) * 64 + h) * 64 + w0;
    out[base + nlo]      = acc0[r];
    out[base + 16 + nlo] = acc1[r];
  }
}

extern "C" void kernel_launch(void* const* d_in, const int* in_sizes, int n_in,
                              void* d_out, int out_size, void* d_ws, size_t ws_size,
                              hipStream_t stream) {
  const float* x     = (const float*)d_in[0];
  const float* w_def = (const float*)d_in[1];
  const float* w_off = (const float*)d_in[2];
  float* out = (float*)d_out;

  float*        offs  = (float*)d_ws;                               // 1,179,648 B
  unsigned int* wfrag = (unsigned int*)((char*)d_ws + OFFS_BYTES);  // 1,179,648 B

  offs_conv  <<<294912 / 256, 256, 0, stream>>>(x, w_off, offs);
  wfrag_build<<<294912 / 256, 256, 0, stream>>>(w_def, wfrag);
  deform_wmma<<<512, 512, 0, stream>>>(x, offs, wfrag, out);
}
